// Cond_LSTM_5334349381924
// MI455X (gfx1250) — matrix operand order repaired and checked by reading, not compiled, not run on hardware
//
#include <hip/hip_runtime.h>
#include <hip/hip_bf16.h>

typedef __attribute__((ext_vector_type(16))) __bf16 v16bf;
typedef __attribute__((ext_vector_type(8)))  __bf16 v8bf;
typedef __attribute__((ext_vector_type(8)))  float  v8f;

#define T_STEP 256
#define NB     64      // batch
#define NI     1024    // input
#define NH     1024    // hidden
#define NF     512
#define NC     512
#define HB     (NH*NB)

// ---- WMMA helpers -----------------------------------------------------------

static __device__ __forceinline__ v8f wmma_bf16(v16bf a, v16bf b, v8f c) {
  return __builtin_amdgcn_wmma_f32_16x16x32_bf16(
      /*neg_a=*/false, a, /*neg_b=*/false, b,
      /*c_mod=*/(short)0, c, /*reuse_a=*/false, /*reuse_b=*/false);
}

// A fragment (16x32 bf16): lane holds row m=lane&15; half=lane>>4 selects k-offset.
// Elements e -> k = k0 + half*16 + e (16 contiguous)
static __device__ __forceinline__ v16bf load_a_frag(const __bf16* p) {
  v8bf lo = *(const v8bf*)(p);
  v8bf hi = *(const v8bf*)(p + 8);
  v16bf a;
#pragma unroll
  for (int e = 0; e < 8; ++e) { a[e] = lo[e]; a[8 + e] = hi[e]; }
  return a;
}

// B fragment (32x16 bf16): lane holds col n=lane&15; k = k0 + half*16 + e (16 contiguous)
static __device__ __forceinline__ v16bf load_b_frag(const __bf16* p) {
  return *(const v16bf*)(p);
}

// ---- Small prep kernels -----------------------------------------------------

__global__ void k_f(const float* __restrict__ F, const float* __restrict__ cond,
                    float* __restrict__ f) {
  int i = blockIdx.x * blockDim.x + threadIdx.x;
  if (i < NF) {
    float s = 0.f;
    for (int k = 0; k < NC; ++k) s += F[i * NC + k] * cond[k];
    f[i] = s;
  }
}

__global__ void k_convert(const float* __restrict__ in, __bf16* __restrict__ out, int n) {
  int i = blockIdx.x * blockDim.x + threadIdx.x;
  if (i < n) out[i] = (__bf16)in[i];
}

// VsT[g][n][k] = f[k] * V[g][k][n]   (N-major for WMMA B fragments)
__global__ void k_build_vst(const float* __restrict__ V, const float* __restrict__ f,
                            __bf16* __restrict__ VsT) {
  int i = blockIdx.x * blockDim.x + threadIdx.x;
  if (i >= 4 * NH * NF) return;
  int k = i % NF;
  int n = (i / NF) % NH;
  int g = i / (NF * NH);
  VsT[i] = (__bf16)(f[k] * V[(g * NF + k) * NH + n]);
}

__global__ void k_init(const float* __restrict__ h0, const float* __restrict__ c0,
                       float* __restrict__ hs0, float* __restrict__ cs0,
                       __bf16* __restrict__ hT0) {
  int i = blockIdx.x * blockDim.x + threadIdx.x;
  if (i >= HB) return;
  int h = i / NB, b = i % NB;
  float hv = h0[i];
  hs0[i] = hv;
  cs0[i] = c0[i];
  hT0[b * NH + h] = (__bf16)hv;   // bf16, batch-major (B fragments need K-contiguous)
}

// ---- WH = U_bf @ VsT^T  (4 GEMMs, M=N=1024, K=512), 32x32 tile per wave ----

__global__ void k_gemm_wh(const __bf16* __restrict__ U, const __bf16* __restrict__ VsT,
                          __bf16* __restrict__ WH) {
  int wave = blockIdx.x * (blockDim.x >> 5) + (threadIdx.x >> 5);
  int lane = threadIdx.x & 31;
  int half = lane >> 4, lr = lane & 15;
  int g   = wave >> 10;          // 1024 tiles per gate (32x32 tiles)
  int rem = wave & 1023;
  int mT  = rem >> 5, nT = rem & 31;

  const __bf16* A0  = U   + ((size_t)g * NH + mT * 32 +  0 + lr) * NF;
  const __bf16* A1  = U   + ((size_t)g * NH + mT * 32 + 16 + lr) * NF;
  const __bf16* Bt0 = VsT + ((size_t)g * NH + nT * 32 +  0 + lr) * NF;
  const __bf16* Bt1 = VsT + ((size_t)g * NH + nT * 32 + 16 + lr) * NF;

  v8f acc00 = {}, acc01 = {}, acc10 = {}, acc11 = {};
  for (int k0 = 0; k0 < NF; k0 += 32) {
    v16bf b0 = load_b_frag(Bt0 + k0 + half * 16);
    v16bf b1 = load_b_frag(Bt1 + k0 + half * 16);
    v16bf a0 = load_a_frag(A0  + k0 + half * 16);
    v16bf a1 = load_a_frag(A1  + k0 + half * 16);
    acc00 = wmma_bf16(a0, b0, acc00);
    acc01 = wmma_bf16(a0, b1, acc01);
    acc10 = wmma_bf16(a1, b0, acc10);
    acc11 = wmma_bf16(a1, b1, acc11);
  }
  __bf16* C = WH + (size_t)g * NH * NH;
#pragma unroll
  for (int r = 0; r < 8; ++r) {
    int m0 = mT * 32 + r + 8 * half;
    int m1 = m0 + 16;
    int n0 = nT * 32 + lr;
    int n1 = n0 + 16;
    C[m0 * NH + n0] = (__bf16)acc00[r];
    C[m0 * NH + n1] = (__bf16)acc01[r];
    C[m1 * NH + n0] = (__bf16)acc10[r];
    C[m1 * NH + n1] = (__bf16)acc11[r];
  }
}

// ---- xproj[t][g][h][b] = WI_bf[g] @ x_bf  (M=4096, N=16384, K=1024) --------
// 64x32 tile per wave: 4 A-fragments x 2 B-fragments, 8 accumulators.

__global__ void k_gemm_xproj(const __bf16* __restrict__ WIb, const __bf16* __restrict__ xb,
                             float* __restrict__ xproj) {
  int wave = blockIdx.x * (blockDim.x >> 5) + (threadIdx.x >> 5);
  int lane = threadIdx.x & 31;
  int half = lane >> 4, lr = lane & 15;
  int mT = wave >> 9;            // 0..63   (64-row tile over M=4096)
  int nT = wave & 511;           // 0..511  (32-col tile over N=16384)
  int g     = mT >> 4;
  int hBase = (mT & 15) * 64;

  const __bf16* A[4];
#pragma unroll
  for (int i = 0; i < 4; ++i)
    A[i] = WIb + ((size_t)g * NH + hBase + i * 16 + lr) * NI;
  const __bf16* Bt0 = xb + ((size_t)nT * 32 +  0 + lr) * NI;
  const __bf16* Bt1 = xb + ((size_t)nT * 32 + 16 + lr) * NI;

  v8f acc[4][2];
#pragma unroll
  for (int i = 0; i < 4; ++i) { acc[i][0] = (v8f){}; acc[i][1] = (v8f){}; }

  for (int k0 = 0; k0 < NI; k0 += 32) {
    v16bf b0 = load_b_frag(Bt0 + k0 + half * 16);
    v16bf b1 = load_b_frag(Bt1 + k0 + half * 16);
#pragma unroll
    for (int i = 0; i < 4; ++i) {
      v16bf a = load_a_frag(A[i] + k0 + half * 16);
      acc[i][0] = wmma_bf16(a, b0, acc[i][0]);
      acc[i][1] = wmma_bf16(a, b1, acc[i][1]);
    }
  }

  int t  = nT >> 1;              // 32-wide n tile stays inside one timestep (32 | 64)
  int b0c = (nT & 1) * 32;
#pragma unroll
  for (int i = 0; i < 4; ++i) {
#pragma unroll
    for (int r = 0; r < 8; ++r) {
      int h = hBase + i * 16 + r + 8 * half;
      size_t base = (((size_t)t * 4 + g) * NH + h) * NB;
      xproj[base + b0c +  0 + lr] = acc[i][0][r];
      xproj[base + b0c + 16 + lr] = acc[i][1][r];
    }
  }
}

// ---- One recurrence step: z = xp + WH@h ; fused gates ----------------------
// 256 waves: 64 h-tiles x 4 b-tiles, each wave does all 4 gates (shared B frag)

__global__ void k_step(const __bf16* __restrict__ WH, const float* __restrict__ xp,
                       const __bf16* __restrict__ hT_in, __bf16* __restrict__ hT_out,
                       const float* __restrict__ cs_t,
                       float* __restrict__ hs_next, float* __restrict__ cs_next) {
  int wave = blockIdx.x * (blockDim.x >> 5) + (threadIdx.x >> 5);
  int lane = threadIdx.x & 31;
  int half = lane >> 4, lr = lane & 15;
  int mT = wave >> 2;            // 0..63
  int nT = wave & 3;             // 0..3

  const __bf16* Bt = hT_in + (nT * 16 + lr) * NH;
  const __bf16* A0 = WH + (size_t)0 * NH * NH + (mT * 16 + lr) * NH;
  const __bf16* A1 = WH + (size_t)1 * NH * NH + (mT * 16 + lr) * NH;
  const __bf16* A2 = WH + (size_t)2 * NH * NH + (mT * 16 + lr) * NH;
  const __bf16* A3 = WH + (size_t)3 * NH * NH + (mT * 16 + lr) * NH;

  v8f acc0 = {}, acc1 = {}, acc2 = {}, acc3 = {};
  for (int k0 = 0; k0 < NH; k0 += 32) {
    v16bf b  = load_b_frag(Bt + k0 + half * 16);
    v16bf a0 = load_a_frag(A0 + k0 + half * 16);
    v16bf a1 = load_a_frag(A1 + k0 + half * 16);
    v16bf a2 = load_a_frag(A2 + k0 + half * 16);
    v16bf a3 = load_a_frag(A3 + k0 + half * 16);
    acc0 = wmma_bf16(a0, b, acc0);
    acc1 = wmma_bf16(a1, b, acc1);
    acc2 = wmma_bf16(a2, b, acc2);
    acc3 = wmma_bf16(a3, b, acc3);
  }

#pragma unroll
  for (int r = 0; r < 8; ++r) {
    int m = mT * 16 + r + 8 * half;
    int n = nT * 16 + lr;
    int idx = m * NB + n;
    float zi = acc0[r] + xp[0 * HB + idx];
    float zf = acc1[r] + xp[1 * HB + idx];
    float zg = acc2[r] + xp[2 * HB + idx];
    float zo = acc3[r] + xp[3 * HB + idx];
    float ig = 1.f / (1.f + __expf(-zi));
    float fg = 1.f / (1.f + __expf(-zf));
    float gg = tanhf(zg);
    float og = 1.f / (1.f + __expf(-zo));
    float cn = fg * cs_t[idx] + ig * gg;
    float hn = og * tanhf(cn);
    cs_next[idx] = cn;
    hs_next[idx] = hn;
    hT_out[n * NH + m] = (__bf16)hn;
  }
}

// ---- Host side --------------------------------------------------------------

extern "C" void kernel_launch(void* const* d_in, const int* in_sizes, int n_in,
                              void* d_out, int out_size, void* d_ws, size_t ws_size,
                              hipStream_t stream) {
  const float* x    = (const float*)d_in[0];   // (256,64,1024)
  const float* h0   = (const float*)d_in[1];   // (1024,64)
  const float* c0   = (const float*)d_in[2];
  const float* cond = (const float*)d_in[3];   // (512,)
  const float* F    = (const float*)d_in[4];   // (512,512)
  const float* U    = (const float*)d_in[5];   // (4,1024,512)
  const float* V    = (const float*)d_in[6];   // (4,512,1024)
  const float* WI   = (const float*)d_in[7];   // (4,1024,1024)

  char* ws = (char*)d_ws;
  size_t off = 0;
  auto alloc = [&](size_t bytes) -> void* {
    void* p = ws + off;
    off = (off + bytes + 255) & ~(size_t)255;
    return p;
  };
  float*  f_d    = (float*) alloc((size_t)NF * 4);
  __bf16* U_bf   = (__bf16*)alloc((size_t)4 * NH * NF * 2);
  __bf16* VsT_bf = (__bf16*)alloc((size_t)4 * NH * NF * 2);
  __bf16* WI_bf  = (__bf16*)alloc((size_t)4 * NH * NI * 2);
  __bf16* x_bf   = (__bf16*)alloc((size_t)T_STEP * NB * NI * 2);
  __bf16* WH_bf  = (__bf16*)alloc((size_t)4 * NH * NH * 2);
  __bf16* hT0    = (__bf16*)alloc((size_t)NB * NH * 2);
  __bf16* hT1    = (__bf16*)alloc((size_t)NB * NH * 2);
  float*  xproj  = (float*) alloc((size_t)T_STEP * 4 * NH * NB * 4);

  float* hs = (float*)d_out;                       // (257,1024,64)
  float* cs = hs + (size_t)(T_STEP + 1) * HB;      // (257,1024,64)

  // 1) hyper-vector f
  k_f<<<(NF + 255) / 256, 256, 0, stream>>>(F, cond, f_d);

  // 2) bf16 conversions / staging
  {
    int n = 4 * NH * NF;
    k_convert<<<(n + 255) / 256, 256, 0, stream>>>(U, U_bf, n);
  }
  {
    int n = 4 * NH * NI;
    k_convert<<<(n + 255) / 256, 256, 0, stream>>>(WI, WI_bf, n);
  }
  {
    int n = T_STEP * NB * NI;
    k_convert<<<(n + 255) / 256, 256, 0, stream>>>(x, x_bf, n);
  }
  {
    int n = 4 * NH * NF;
    k_build_vst<<<(n + 255) / 256, 256, 0, stream>>>(V, f_d, VsT_bf);
  }

  // 3) WH = U @ (diag(f) V)    : 4096 waves (32x32 tiles)
  k_gemm_wh<<<512, 256, 0, stream>>>(U_bf, VsT_bf, WH_bf);

  // 4) xproj = WI @ x for all timesteps : 32768 waves (64x32 tiles)
  k_gemm_xproj<<<4096, 256, 0, stream>>>(WI_bf, x_bf, xproj);

  // 5) state init
  k_init<<<(HB + 255) / 256, 256, 0, stream>>>(h0, c0, hs, cs, hT0);

  // 6) 256 sequential recurrence steps (ping-pong bf16 hT buffers)
  for (int t = 0; t < T_STEP; ++t) {
    __bf16* hin  = (t & 1) ? hT1 : hT0;
    __bf16* hout = (t & 1) ? hT0 : hT1;
    k_step<<<32, 256, 0, stream>>>(WH_bf,
                                   xproj + (size_t)t * 4 * HB,
                                   hin, hout,
                                   cs + (size_t)t * HB,
                                   hs + (size_t)(t + 1) * HB,
                                   cs + (size_t)(t + 1) * HB);
  }
}
